// SUnetDecoder_21577915695668
// MI455X (gfx1250) — compile-verified, matrix operand order checked by reading, not run on hardware
//
#include <hip/hip_runtime.h>

typedef unsigned short u16;
typedef unsigned int   u32;
typedef unsigned long long u64;
typedef __attribute__((ext_vector_type(16))) __bf16 v16bf;
typedef __attribute__((ext_vector_type(8)))  float  v8f;
typedef __attribute__((ext_vector_type(4)))  u32    u32x4;
typedef __attribute__((ext_vector_type(8)))  u32    u32x8;

#define DIM   128
#define DH    64
#define HH    56
#define WWID  56
#define NTOK  3136            // 56*56
#define BATCH 16
#define MTOK  (BATCH*NTOK)    // 50176
#define NP    196             // 14*14
#define HID   512

// ---------------- scalar bf16 helpers (manual RNE, no header dependence) ----
__device__ __forceinline__ u16 f2bf(float f) {
  u32 u = __builtin_bit_cast(u32, f);
  u32 r = 0x7fffu + ((u >> 16) & 1u);
  return (u16)((u + r) >> 16);
}
__device__ __forceinline__ float bf2f(u16 h) {
  u32 u = ((u32)h) << 16;
  return __builtin_bit_cast(float, u);
}
__device__ __forceinline__ u32 hash_u32(u32 x) {
  x ^= x >> 16; x *= 0x7feb352dU; x ^= x >> 15; x *= 0x846ca68bU; x ^= x >> 16;
  return x;
}
__device__ __forceinline__ v8f vzero8() { v8f z; for (int i = 0; i < 8; ++i) z[i] = 0.f; return z; }

__device__ __forceinline__ v8f wmma_bf16(v16bf a, v16bf b, v8f c) {
  return __builtin_amdgcn_wmma_f32_16x16x32_bf16(false, a, false, b, (short)0, c, false, false);
}

// A (or weight-style B) fragment, 16x32 bf16 from row-major [rows][stride].
// Lane l: row = row0 + (l&15) (clamped); k pattern = (i<8?0:16) + (l>=16?8:0) + (i&7).
__device__ __forceinline__ v16bf frag_rowmajor(const u16* __restrict__ base,
                                               int row0, int rowMax, int stride, int kOff) {
  int l = threadIdx.x & 31;
  int r = row0 + (l & 15);
  if (r > rowMax) r = rowMax;
  const u16* p = base + (size_t)r * stride + kOff + ((l >> 4) << 3);
  union { uint4 q[2]; v16bf v; } u;
  u.q[0] = *(const uint4*)(p);
  u.q[1] = *(const uint4*)(p + 16);
  return u.v;
}
// Same pattern from an LDS tile (stride must be multiple of 8 elements).
__device__ __forceinline__ v16bf frag_lds16(const u16* base, int stride) {
  int l = threadIdx.x & 31;
  const u16* p = base + (l & 15) * stride + ((l >> 4) << 3);
  union { uint4 q[2]; v16bf v; } u;
  u.q[0] = *(const uint4*)(p);
  u.q[1] = *(const uint4*)(p + 16);
  return u.v;
}

// ---- Tensor Data Mover: 2D tile [rows x 32 bf16] global -> LDS -------------
// D# per CDNA5 ISA. pad: every 16 DWORDs (64B row) insert
// 4 DWORDs (16B) -> LDS row stride 40 u16 elements (bank-conflict free).
// ldsDst is passed into the asm (unreferenced) so the LDS object escapes and
// the compiler treats the DMA as a writer of that memory (memory clobber).
__device__ __forceinline__ void tdm_load_tile(const u16* W, u32 rowBase, u32 rowsAvail,
                                              u32 K, u32 kb, u32 ldsByteOff,
                                              void* ldsDst) {
  u64 ga = (u64)(size_t)W + ((u64)rowBase * (u64)K + (u64)kb) * 2ull;
  u32x4 g0;
  g0[0] = 1u;                                   // count=1, user mode, no gather
  g0[1] = ldsByteOff;                           // lds_addr (bytes)
  g0[2] = (u32)ga;                              // global_addr[31:0]
  g0[3] = (u32)((ga >> 32) & 0x1FFFFFFu) | (2u << 30);  // addr[56:32] | type=2
  u32x8 g1;
  g1[0] = (1u << 16)                            // data_size = 2 bytes
        | (1u << 20)                            // pad_enable
        | (3u << 22)                            // pad_interval: 16 DWORDs
        | (3u << 25);                           // pad_amount:   4 DWORDs
  g1[1] = (K & 0xFFFFu) << 16;                  // tensor_dim0[15:0] in [31:16]
  g1[2] = (K >> 16) | ((rowsAvail & 0xFFFFu) << 16);    // dim0 hi | dim1 lo
  g1[3] = (rowsAvail >> 16) | (32u << 16);      // dim1 hi | tile_dim0 = 32
  g1[4] = 64u;                                  // tile_dim1 = 64, tile_dim2 = 0
  g1[5] = K;                                    // tensor_dim0_stride[31:0]
  g1[6] = 0u;                                   // stride hi | dim1_stride lo
  g1[7] = 0u;
  asm volatile("tensor_load_to_lds %0, %1"
               :: "s"(g0), "s"(g1), "v"(ldsDst)
               : "memory");
}

// ---------------- generic bf16 WMMA GEMM ------------------------------------
// C[M x N] = A[M x K] * W[N x K]^T (+bias).  mode: 0=f32 store, 1=bf16 store,
// 2=f32 residual accumulate.  Block: 256 thr; tile M=128 (8 waves x 16), N=64.
// B tiles staged into LDS by the Tensor Data Mover, double buffered.
__global__ __launch_bounds__(256) void gemm_bf16(
    const u16* __restrict__ A, const u16* __restrict__ W,
    const float* __restrict__ bias, float* __restrict__ outF,
    u16* __restrict__ outB, int M, int N, int K, int mode)
{
  __shared__ u16 bs[2][64 * 40];        // [n][k] tiles, stride 40 (via TDM pad)
  // Never executes: guarantees the LDS object has a store in IR (the real
  // writer is the TDM, which the compiler cannot see).
  if (blockDim.x == 0) bs[0][threadIdx.x] = 0;

  int nBase = blockIdx.x * 64;
  int mBase = blockIdx.y * 128;
  int t  = threadIdx.x;
  int wv = t >> 5;
  int l  = t & 31;
  v8f acc[4];
  for (int i = 0; i < 4; ++i) acc[i] = vzero8();
  int row0 = mBase + wv * 16;
  int rowA = row0 < M ? row0 : (M - 16 > 0 ? M - 16 : 0);

  if (wv == 0)                          // preload first B tile (wave-level DMA)
    tdm_load_tile(W, (u32)nBase, (u32)(N - nBase), (u32)K, 0u, 0u, &bs[0][0]);

  int buf = 0;
  for (int kb = 0; kb < K; kb += 32) {
    if (wv == 0) {
      if (kb + 32 < K) {                // issue next tile, wait for current
        tdm_load_tile(W, (u32)nBase, (u32)(N - nBase), (u32)K, (u32)(kb + 32),
                      (u32)((buf ^ 1) * 64 * 40 * 2), &bs[buf ^ 1][0]);
        __builtin_amdgcn_s_wait_tensorcnt(1);
      } else {
        __builtin_amdgcn_s_wait_tensorcnt(0);
      }
    }
    __syncthreads();                    // B tile visible to all waves
    if (kb + 32 < K)
      __builtin_prefetch(A + (size_t)rowA * K + kb + 32, 0, 1);   // global_prefetch_b8
    v16bf af = frag_rowmajor(A, row0, M - 1, K, kb);
#pragma unroll
    for (int tn = 0; tn < 4; ++tn) {
      v16bf bf = frag_lds16(&bs[buf][(tn * 16) * 40], 40);
      acc[tn] = wmma_bf16(af, bf, acc[tn]);
    }
    __syncthreads();                    // all reads done before DMA reuses buf
    buf ^= 1;
  }
  int rhi = (l >> 4) << 3;
#pragma unroll
  for (int tn = 0; tn < 4; ++tn) {
    int col = nBase + tn * 16 + (l & 15);
    float bv = bias ? bias[col] : 0.f;
#pragma unroll
    for (int r = 0; r < 8; ++r) {
      int row = row0 + r + rhi;
      if (row < M) {
        float v = acc[tn][r] + bv;
        size_t o = (size_t)row * N + col;
        if (mode == 0)      outF[o] = v;
        else if (mode == 1) outB[o] = f2bf(v);
        else                outF[o] += v;
      }
    }
  }
}

// ---------------- spatial-reduction conv as implicit GEMM -------------------
// xr[3136 rows (b,np)][128] = im2col(h_bf 4x4 stride4) * sr_w^T + sr_b
// Block 256 thr: M tile 32 (2 row-groups), N=128 (waves: 2M x 4N, 16x32 each).
__global__ __launch_bounds__(256) void srconv_bf16(
    const u16* __restrict__ hbf, const u16* __restrict__ wsr,
    const float* __restrict__ bias, float* __restrict__ xr)
{
  __shared__ u16 bs[128 * 40];
  int mBase = blockIdx.x * 32;
  int t = threadIdx.x, wv = t >> 5, l = t & 31;
  int mw = mBase + (wv >> 2) * 16;      // row-group of this wave
  int nq = (wv & 3) * 32;               // 32 output channels of this wave
  v8f acc[2]; acc[0] = vzero8(); acc[1] = vzero8();

  int m  = mw + (l & 15);               // always < 3136 (98 blocks * 32)
  int b  = m / NP;
  int np = m - b * NP;
  int npy = np / 14, npx = np - npy * 14;

  for (int kb = 0; kb < 2048; kb += 32) {
    int p = kb >> 7;                    // patch position 0..15 (ky*4+kx)
    int ci0 = kb & 127;
    int ky = p >> 2, kx = p & 3;
    for (int idx = t; idx < 4096; idx += 256) {    // stage W [128 co][32 ci]
      int n = idx >> 5, kk = idx & 31;
      bs[n * 40 + kk] = wsr[((size_t)n * 128 + ci0 + kk) * 16 + p];
    }
    __syncthreads();
    int ny = npy * 4 + ky, nx = npx * 4 + kx;
    const u16* ap = hbf + ((size_t)b * NTOK + ny * WWID + nx) * DIM
                        + ci0 + ((l >> 4) << 3);
    union { uint4 q[2]; v16bf v; } ua;
    ua.q[0] = *(const uint4*)(ap);
    ua.q[1] = *(const uint4*)(ap + 16);
#pragma unroll
    for (int tn = 0; tn < 2; ++tn) {
      v16bf bf = frag_lds16(&bs[(nq + tn * 16) * 40], 40);
      acc[tn] = wmma_bf16(ua.v, bf, acc[tn]);
    }
    __syncthreads();
  }
  int rhi = (l >> 4) << 3;
#pragma unroll
  for (int tn = 0; tn < 2; ++tn) {
    int col = nq + tn * 16 + (l & 15);
    float bv = bias[col];
#pragma unroll
    for (int r = 0; r < 8; ++r) {
      int row = mw + r + rhi;
      xr[(size_t)row * DIM + col] = acc[tn][r] + bv;
    }
  }
}

// ---------------- fused SR attention ----------------------------------------
// grid (B*2 heads, 25 q-tiles); block 256 (8 waves x 16 q rows). Np=196 keys.
__global__ __launch_bounds__(256) void attn_fused(
    const u16* __restrict__ qbf, const u16* __restrict__ kvbf,
    u16* __restrict__ obf, u32 seed)
{
  __shared__ u16 plds[8 * 16 * 232];    // per-wave P tile [16][232] bf16
  int bh = blockIdx.x;
  int b = bh >> 1, head = bh & 1;
  int wv = threadIdx.x >> 5, l = threadIdx.x & 31;
  int q0 = blockIdx.y * 128 + wv * 16;  // query row base within batch b
  int rhi = (l >> 4) << 3;

  const u16* qbase = qbf + (size_t)b * NTOK * DIM + head * DH;
  v16bf qf0 = frag_rowmajor(qbase, q0, NTOK - 1, DIM, 0);
  v16bf qf1 = frag_rowmajor(qbase, q0, NTOK - 1, DIM, 32);

  // ---- S = Q K^T : 13 tiles of 16 keys (196 -> 208 with masking) ----
  v8f s[13];
#pragma unroll
  for (int j = 0; j < 13; ++j) s[j] = vzero8();
#pragma unroll
  for (int j = 0; j < 13; ++j) {
    int key = j * 16 + (l & 15); if (key > 195) key = 195;
    const u16* kp = kvbf + ((size_t)b * NP + key) * 256 + head * DH + rhi;
    union { uint4 q[2]; v16bf v; } uk;
    uk.q[0] = *(const uint4*)(kp);      uk.q[1] = *(const uint4*)(kp + 16);
    s[j] = wmma_bf16(qf0, uk.v, s[j]);
    uk.q[0] = *(const uint4*)(kp + 32); uk.q[1] = *(const uint4*)(kp + 48);
    s[j] = wmma_bf16(qf1, uk.v, s[j]);
  }

  // ---- scale + dropkey + masked softmax (row across 16-lane half) ----
#pragma unroll
  for (int r = 0; r < 8; ++r) {
    int row = q0 + r + rhi;
    float mx = -3.0e38f;
#pragma unroll
    for (int j = 0; j < 13; ++j) {
      int col = j * 16 + (l & 15);
      float v = s[j][r] * 0.125f;       // dh^-0.5, dh=64
      if (col >= 196) v = -3.0e38f;
      else {
        u32 idx = (((u32)bh * 3200u + (u32)row) * 224u + (u32)col) ^ seed;
        if (hash_u32(idx) < 429496730u) v -= 1e12f;   // p = 0.1
      }
      s[j][r] = v;
      mx = fmaxf(mx, v);
    }
#pragma unroll
    for (int off = 1; off < 16; off <<= 1) mx = fmaxf(mx, __shfl_xor(mx, off, 32));
    float sum = 0.f;
#pragma unroll
    for (int j = 0; j < 13; ++j) { float e = __expf(s[j][r] - mx); s[j][r] = e; sum += e; }
#pragma unroll
    for (int off = 1; off < 16; off <<= 1) sum += __shfl_xor(sum, off, 32);
    float inv = 1.0f / sum;
#pragma unroll
    for (int j = 0; j < 13; ++j) s[j][r] *= inv;
  }

  // ---- P -> LDS (bf16, A-layout-readable), pad cols 208..223 with 0 ----
  u16* my = plds + wv * (16 * 232);
#pragma unroll
  for (int j = 0; j < 13; ++j)
#pragma unroll
    for (int r = 0; r < 8; ++r)
      my[(r + rhi) * 232 + j * 16 + (l & 15)] = f2bf(s[j][r]);
#pragma unroll
  for (int r = 0; r < 8; ++r)
    my[(r + rhi) * 232 + 208 + (l & 15)] = 0;
  __syncthreads();

  // ---- O = P V : 7 k-chunks of 32 keys, 4 d-tiles of 16 ----
  v8f o[4];
#pragma unroll
  for (int i = 0; i < 4; ++i) o[i] = vzero8();
  for (int kc = 0; kc < 7; ++kc) {
    v16bf pf = frag_lds16(my + kc * 32, 232);
#pragma unroll
    for (int d4 = 0; d4 < 4; ++d4) {
      union { u16 e[16]; v16bf v; } uv;
      int d = d4 * 16 + (l & 15);
#pragma unroll
      for (int i = 0; i < 16; ++i) {
        int key = kc * 32 + ((i < 8) ? 0 : 16) + rhi + (i & 7);
        if (key > 195) key = 195;       // P==0 there
        uv.e[i] = kvbf[((size_t)b * NP + key) * 256 + 128 + head * DH + d];
      }
      o[d4] = wmma_bf16(pf, uv.v, o[d4]);
    }
  }
#pragma unroll
  for (int d4 = 0; d4 < 4; ++d4)
#pragma unroll
    for (int r = 0; r < 8; ++r) {
      int row = q0 + r + rhi;
      if (row < NTOK)
        obf[((size_t)b * NTOK + row) * DIM + head * DH + d4 * 16 + (l & 15)] =
            f2bf(o[d4][r]);
    }
}

// ---------------- LayerNorm f32 -> bf16, 128 channels -----------------------
__global__ __launch_bounds__(256) void ln_bf16(
    const float* __restrict__ x, const float* __restrict__ g,
    const float* __restrict__ bta, u16* __restrict__ out, int M)
{
  int row = blockIdx.x * 8 + (threadIdx.x >> 5);
  int l = threadIdx.x & 31;
  if (row >= M) return;
  const float* xr = x + (size_t)row * DIM;
  float v[4], s = 0.f, s2 = 0.f;
#pragma unroll
  for (int i = 0; i < 4; ++i) { v[i] = xr[l + 32 * i]; s += v[i]; s2 += v[i] * v[i]; }
#pragma unroll
  for (int off = 1; off < 32; off <<= 1) { s += __shfl_xor(s, off, 32); s2 += __shfl_xor(s2, off, 32); }
  float mean = s * (1.f / 128.f);
  float var  = s2 * (1.f / 128.f) - mean * mean;
  float inv  = rsqrtf(var + 1e-5f);
#pragma unroll
  for (int i = 0; i < 4; ++i) {
    int c = l + 32 * i;
    out[(size_t)row * DIM + c] = f2bf((v[i] - mean) * inv * g[c] + bta[c]);
  }
}

// ---------------- depthwise 3x3 SAME + exact GELU (bf16 in/out) -------------
__global__ __launch_bounds__(256) void dwgelu(
    const u16* __restrict__ f1, const float* __restrict__ wdw,
    const float* __restrict__ bdw, u16* __restrict__ out)
{
  size_t idx = (size_t)blockIdx.x * 256 + threadIdx.x;   // B*N*HID
  int c = (int)(idx & 511);
  size_t bn = idx >> 9;
  int n = (int)(bn % NTOK);
  int b = (int)(bn / NTOK);
  int h = n / WWID, w = n - h * WWID;
  float acc = bdw[c];
#pragma unroll
  for (int dy = -1; dy <= 1; ++dy)
#pragma unroll
    for (int dx = -1; dx <= 1; ++dx) {
      int hh = h + dy, ww = w + dx;
      if (hh >= 0 && hh < HH && ww >= 0 && ww < WWID)
        acc += bf2f(f1[((size_t)b * NTOK + hh * WWID + ww) * HID + c])
               * wdw[c * 9 + (dy + 1) * 3 + (dx + 1)];
    }
  float ge = 0.5f * acc * (1.f + erff(acc * 0.70710678118654752f));
  out[idx] = f2bf(ge);
}

// ---------------- NCHW f32 -> [B,N,C] bf16 (concat input) -------------------
__global__ void pack_x(const float* __restrict__ x, u16* __restrict__ xt) {
  __shared__ float tile[32][33];
  int n0 = blockIdx.x * 32, c0 = blockIdx.y * 32, b = blockIdx.z;
  int tx = threadIdx.x, ty = threadIdx.y;
#pragma unroll
  for (int i = 0; i < 4; ++i)
    tile[ty + 8 * i][tx] = x[((size_t)b * 256 + c0 + ty + 8 * i) * NTOK + n0 + tx];
  __syncthreads();
#pragma unroll
  for (int i = 0; i < 4; ++i)
    xt[((size_t)b * NTOK + n0 + ty + 8 * i) * 256 + c0 + tx] = f2bf(tile[tx][ty + 8 * i]);
}

// ---------------- [B,N,128] f32 -> NCHW f32 output --------------------------
__global__ void unpack_x(const float* __restrict__ xs, float* __restrict__ out) {
  __shared__ float tile[32][33];
  int n0 = blockIdx.x * 32, c0 = blockIdx.y * 32, b = blockIdx.z;
  int tx = threadIdx.x, ty = threadIdx.y;
#pragma unroll
  for (int i = 0; i < 4; ++i)
    tile[ty + 8 * i][tx] = xs[((size_t)b * NTOK + n0 + ty + 8 * i) * DIM + c0 + tx];
  __syncthreads();
#pragma unroll
  for (int i = 0; i < 4; ++i)
    out[((size_t)b * DIM + c0 + ty + 8 * i) * NTOK + n0 + tx] = tile[tx][ty + 8 * i];
}

__global__ void cvt_f32_bf16(const float* __restrict__ s, u16* __restrict__ d, int n) {
  int i = blockIdx.x * 256 + threadIdx.x;
  if (i < n) d[i] = f2bf(s[i]);
}

// ============================================================================
extern "C" void kernel_launch(void* const* d_in, const int* in_sizes, int n_in,
                              void* d_out, int out_size, void* d_ws, size_t ws_size,
                              hipStream_t stream)
{
  (void)in_sizes; (void)n_in; (void)out_size; (void)ws_size;
  const float* x        = (const float*)d_in[0];
  const float* concat_w = (const float*)d_in[1];
  const float* concat_b = (const float*)d_in[2];
  const float* norm1_g  = (const float*)d_in[3];
  const float* norm1_b  = (const float*)d_in[4];
  const float* q_w      = (const float*)d_in[5];
  const float* kv_w     = (const float*)d_in[6];
  const float* sr_w     = (const float*)d_in[7];
  const float* sr_b     = (const float*)d_in[8];
  const float* srn_g    = (const float*)d_in[9];
  const float* srn_b    = (const float*)d_in[10];
  const float* proj_w   = (const float*)d_in[11];
  const float* proj_b   = (const float*)d_in[12];
  const float* norm2_g  = (const float*)d_in[13];
  const float* norm2_b  = (const float*)d_in[14];
  const float* fc1_w    = (const float*)d_in[15];
  const float* fc1_b    = (const float*)d_in[16];
  const float* dw_w     = (const float*)d_in[17];
  const float* dw_b     = (const float*)d_in[18];
  const float* fc2_w    = (const float*)d_in[19];
  const float* fc2_b    = (const float*)d_in[20];

  char* ws = (char*)d_ws;
  size_t off = 0;
  auto take = [&](size_t bytes) -> char* {
    char* p = ws + off; off = (off + bytes + 255) & ~(size_t)255; return p;
  };
  float* xs   = (float*)take((size_t)MTOK * DIM * 4);
  u16*   R1   = (u16*)  take((size_t)MTOK * HID * 2);   // xt / fc1-out
  u16*   R2   = (u16*)  take((size_t)MTOK * HID * 2);   // h,q,o,h2 / gelu-out
  float* xr   = (float*)take((size_t)3136 * DIM * 4);
  u16*   xrbf = (u16*)  take((size_t)3136 * DIM * 2);
  u16*   kvbf = (u16*)  take((size_t)3136 * 256 * 2);
  u16*   wcat = (u16*)  take((size_t)32768 * 2);
  u16*   wq   = (u16*)  take((size_t)2 * 16384 * 2);
  u16*   wkv  = (u16*)  take((size_t)2 * 32768 * 2);
  u16*   wsr  = (u16*)  take((size_t)2 * 262144 * 2);
  u16*   wpr  = (u16*)  take((size_t)2 * 16384 * 2);
  u16*   wf1  = (u16*)  take((size_t)2 * 65536 * 2);
  u16*   wf2  = (u16*)  take((size_t)2 * 65536 * 2);

  u16* xt  = R1;
  u16* f1o = R1;
  u16* hbf = R2;
  u16* qb  = R2 + (size_t)MTOK * 128;
  u16* ob  = R2 + (size_t)MTOK * 256;
  u16* h2  = R2 + (size_t)MTOK * 384;
  u16* gb  = R2;                         // reused after h/q/o/h2 are dead

  auto cvt = [&](const float* s, u16* d, int n) {
    cvt_f32_bf16<<<(n + 255) / 256, 256, 0, stream>>>(s, d, n);
  };
  cvt(concat_w, wcat, 32768);
  for (int i = 0; i < 2; ++i) {
    cvt(q_w   + (size_t)i * 16384,  wq  + (size_t)i * 16384,  16384);
    cvt(kv_w  + (size_t)i * 32768,  wkv + (size_t)i * 32768,  32768);
    cvt(sr_w  + (size_t)i * 262144, wsr + (size_t)i * 262144, 262144);
    cvt(proj_w+ (size_t)i * 16384,  wpr + (size_t)i * 16384,  16384);
    cvt(fc1_w + (size_t)i * 65536,  wf1 + (size_t)i * 65536,  65536);
    cvt(fc2_w + (size_t)i * 65536,  wf2 + (size_t)i * 65536,  65536);
  }

  pack_x<<<dim3(98, 8, 16), dim3(32, 8), 0, stream>>>(x, xt);

  // xs = xt @ concat_w^T + concat_b
  gemm_bf16<<<dim3(2, 392), 256, 0, stream>>>(xt, wcat, concat_b, xs, nullptr,
                                              MTOK, 128, 256, 0);

  for (int i = 0; i < 2; ++i) {
    ln_bf16<<<(MTOK + 7) / 8, 256, 0, stream>>>(xs, norm1_g + i * 128,
                                                norm1_b + i * 128, hbf, MTOK);
    // q = h @ q_w^T
    gemm_bf16<<<dim3(2, 392), 256, 0, stream>>>(hbf, wq + (size_t)i * 16384,
        nullptr, nullptr, qb, MTOK, 128, 128, 1);
    // spatial reduction conv (4x4 stride 4) + bias
    srconv_bf16<<<98, 256, 0, stream>>>(hbf, wsr + (size_t)i * 262144,
                                        sr_b + i * 128, xr);
    ln_bf16<<<(3136 + 7) / 8, 256, 0, stream>>>(xr, srn_g + i * 128,
                                                srn_b + i * 128, xrbf, 3136);
    // kv = xr @ kv_w^T
    gemm_bf16<<<dim3(4, 25), 256, 0, stream>>>(xrbf, wkv + (size_t)i * 32768,
        nullptr, nullptr, kvbf, 3136, 256, 128, 1);
    // fused attention (dropkey + softmax + PV)
    attn_fused<<<dim3(32, 25), 256, 0, stream>>>(qb, kvbf, ob,
        0x12345u + (u32)i * 0x9e3779b9u);
    // xs += o @ proj_w^T + proj_b
    gemm_bf16<<<dim3(2, 392), 256, 0, stream>>>(ob, wpr + (size_t)i * 16384,
        proj_b + i * 128, xs, nullptr, MTOK, 128, 128, 2);
    // MLP
    ln_bf16<<<(MTOK + 7) / 8, 256, 0, stream>>>(xs, norm2_g + i * 128,
                                                norm2_b + i * 128, h2, MTOK);
    gemm_bf16<<<dim3(8, 392), 256, 0, stream>>>(h2, wf1 + (size_t)i * 65536,
        fc1_b + i * 512, nullptr, f1o, MTOK, 512, 128, 1);
    dwgelu<<<100352, 256, 0, stream>>>(f1o, dw_w + (size_t)i * 512 * 9,
                                       dw_b + i * 512, gb);
    gemm_bf16<<<dim3(2, 392), 256, 0, stream>>>(gb, wf2 + (size_t)i * 65536,
        fc2_b + i * 128, xs, nullptr, MTOK, 128, 512, 2);
  }

  unpack_x<<<dim3(98, 4, 16), dim3(32, 8), 0, stream>>>(xs, (float*)d_out);
}
